// Model_72980084293793
// MI455X (gfx1250) — matrix operand order repaired and checked by reading, not compiled, not run on hardware
//
#include <hip/hip_runtime.h>

typedef __bf16 bf16;
typedef __attribute__((ext_vector_type(16))) __bf16       v16bf;
typedef __attribute__((ext_vector_type(8)))  float        v8f;
typedef __attribute__((ext_vector_type(4)))  float        v4f;
typedef __attribute__((ext_vector_type(4)))  __bf16       v4bf;
typedef __attribute__((ext_vector_type(4)))  unsigned int v4u;
typedef __attribute__((ext_vector_type(8)))  unsigned int v8u;

union FragU { v16bf v; v4u u[2]; };

// ---------------------------------------------------------------------------
// 128x128 block tile, 8 waves (wave32) in 2(M) x 4(N); each wave 64x32 tile
// = 4x2 v_wmma_f32_16x16x32_bf16 accumulators. K-step 32.
// Tiles arrive in LDS via the Tensor Data Mover (tensor_load_to_lds), double
// buffered; TDM hardware padding (16-DWORD interval, 4-DWORD pad) produces an
// LDS row stride of 40 bf16 (80B) so the 16 rows of a fragment hit distinct
// bank groups.
// ---------------------------------------------------------------------------
constexpr int BM = 128, BN = 128, BK = 32, LDSS = 40;

__global__ __launch_bounds__(256) void cvt_bf16_kernel(
    const float* __restrict__ in, bf16* __restrict__ out, int n4) {
  int i = blockIdx.x * blockDim.x + threadIdx.x;
  if (i >= n4) return;
  v4f x = reinterpret_cast<const v4f*>(in)[i];
  v4bf y;
#pragma unroll
  for (int j = 0; j < 4; ++j) y[j] = (bf16)x[j];
  reinterpret_cast<v4bf*>(out)[i] = y;
}

__global__ __launch_bounds__(256) void binarize_bf16_kernel(
    const float* __restrict__ in, bf16* __restrict__ out, int n4) {
  int i = blockIdx.x * blockDim.x + threadIdx.x;
  if (i >= n4) return;
  v4f x = reinterpret_cast<const v4f*>(in)[i];
  v4bf y;
#pragma unroll
  for (int j = 0; j < 4; ++j) y[j] = (x[j] >= 0.0f) ? (bf16)1.0f : (bf16)-1.0f;
  reinterpret_cast<v4bf*>(out)[i] = y;
}

// Workgroup-LDS byte offset of a __shared__ object (AS3 pointers are 32-bit).
__device__ __forceinline__ unsigned lds_off_u32(void* p) {
  return (unsigned)(unsigned long long)(__attribute__((address_space(3))) void*)p;
}

// Issue one TDM load of a (tile_dim1=128 rows) x (tile_dim0=32 elems) bf16
// tile from a row-major tensor with stride k_dim into LDS at lds_off, with
// padding 4 DWORDs after every 16 DWORDs (row stride 80B = 40 bf16).
// D# packing per CDNA5 ISA 8.3/8.4 (2D tensor: groups 2/3 omitted).
__device__ __forceinline__ void tdm_load_tile(unsigned lds_off,
                                              unsigned long long ga,
                                              unsigned k_dim, unsigned rows) {
  v4u g0;
  g0[0] = 1u;                                    // count=1, user descriptor
  g0[1] = lds_off;                               // lds_addr
  g0[2] = (unsigned)ga;                          // global_addr[31:0]
  g0[3] = ((unsigned)(ga >> 32) & 0x01FFFFFFu)   // global_addr[56:32]
          | (2u << 30);                          // type=2 ("image")
  v8u g1;
  g1[0] = (1u << 16)     // data_size = 1 -> 2 bytes
        | (1u << 20)     // pad_enable
        | (3u << 22)     // pad_interval: 3 -> every 16 DWORDs (one 64B row)
        | (3u << 25);    // pad_amount:   3 -> 4 DWORDs (16B) of padding
  g1[1] = (k_dim & 0xFFFFu) << 16;                       // tensor_dim0[15:0]
  g1[2] = (k_dim >> 16) | ((rows & 0xFFFFu) << 16);      // td0[31:16], td1[15:0]
  g1[3] = ((rows >> 16) & 0xFFFFu) | (32u << 16);        // td1[31:16], tile_dim0=32
  g1[4] = 128u;                                          // tile_dim1=128, tile_dim2=0
  g1[5] = k_dim;                                         // tensor_dim0_stride[31:0]
  g1[6] = 0u;                                            // stride0[47:32], stride1 lo
  g1[7] = 0u;
  asm volatile("tensor_load_to_lds %0, %1" ::"s"(g0), "s"(g1) : "memory");
}

// C[M,N] = A[M,K] * W[N,K]^T + bias, optional ReLU, out bf16 or f32.
// A,W row-major bf16. M,N,K multiples of 128.
template <bool RELU, bool OUTBF16>
__global__ __launch_bounds__(256) void gemm_wmma_bf16(
    const bf16* __restrict__ A, const bf16* __restrict__ W,
    const float* __restrict__ bias, bf16* __restrict__ outb,
    float* __restrict__ outf, int M, int N, int K) {
  // [buf][A/W][tile]: 2 * 2 * 10240B = 40KB of the 320KB WGP LDS
  __shared__ __align__(16) bf16 sT[2][2][BM * LDSS];

  const int tid  = threadIdx.x;
  const int m0   = blockIdx.y * BM;
  const int n0   = blockIdx.x * BN;
  const int lane = tid & 31;
  const int wid  = tid >> 5;
  const int wm   = wid >> 2;   // 0..1  -> 64-row half
  const int wn   = wid & 3;    // 0..3  -> 32-col quarter
  const int lr   = lane & 15;
  const int hi   = lane >> 4;

  const unsigned ldsA0 = lds_off_u32(&sT[0][0][0]);
  const unsigned ldsW0 = lds_off_u32(&sT[0][1][0]);
  const unsigned ldsA1 = lds_off_u32(&sT[1][0][0]);
  const unsigned ldsW1 = lds_off_u32(&sT[1][1][0]);
  const unsigned long long baseA =
      (unsigned long long)A + (unsigned long long)m0 * (unsigned)K * 2ull;
  const unsigned long long baseW =
      (unsigned long long)W + (unsigned long long)n0 * (unsigned)K * 2ull;

  // Uniform branch (readfirstlane) so only wave 0 issues TDM ops — the TDM
  // ignores EXEC, so a lane-masked path would issue from every wave.
  const int wid_s = __builtin_amdgcn_readfirstlane(wid);

  v8f acc[4][2];
#pragma unroll
  for (int i = 0; i < 4; ++i)
#pragma unroll
    for (int j = 0; j < 2; ++j) acc[i][j] = (v8f){0, 0, 0, 0, 0, 0, 0, 0};

  const int nsteps = K / BK;

  if (wid_s == 0) {  // prologue: stage 0 into buffer 0
    tdm_load_tile(ldsA0, baseA, (unsigned)K, (unsigned)M);
    tdm_load_tile(ldsW0, baseW, (unsigned)K, (unsigned)N);
  }

  for (int i = 0; i < nsteps; ++i) {
    const int buf = i & 1;
    if (wid_s == 0) {
      if (i + 1 < nsteps) {  // issue next stage into the other buffer
        unsigned long long koff = (unsigned long long)(i + 1) * BK * 2ull;
        tdm_load_tile(buf ? ldsA0 : ldsA1, baseA + koff, (unsigned)K,
                      (unsigned)M);
        tdm_load_tile(buf ? ldsW0 : ldsW1, baseW + koff, (unsigned)K,
                      (unsigned)N);
        __builtin_amdgcn_s_wait_tensorcnt(2);  // stage i's two loads done
      } else {
        __builtin_amdgcn_s_wait_tensorcnt(0);
      }
    }
    __syncthreads();  // LDS buffer `buf` now holds stage i tiles

    const bf16* pA = &sT[buf][0][0];
    const bf16* pW = &sT[buf][1][0];

    FragU a[4], b[2];
#pragma unroll
    for (int mf = 0; mf < 4; ++mf) {
      // A 16x32: lanes 0-15: K=0..7 & 16..23 ; lanes 16-31: K=8..15 & 24..31
      int off = (wm * 64 + mf * 16 + lr) * LDSS + hi * 8;
      a[mf].u[0] = *reinterpret_cast<const v4u*>(&pA[off]);
      a[mf].u[1] = *reinterpret_cast<const v4u*>(&pA[off + 16]);
    }
#pragma unroll
    for (int nf = 0; nf < 2; ++nf) {
      // B 32x16: lanes 0-15: K=0..7 & 16..23 ; lanes 16-31: K=8..15 & 24..31
      int off = (wn * 32 + nf * 16 + lr) * LDSS + hi * 8;
      b[nf].u[0] = *reinterpret_cast<const v4u*>(&pW[off]);
      b[nf].u[1] = *reinterpret_cast<const v4u*>(&pW[off + 16]);
    }

#pragma unroll
    for (int mf = 0; mf < 4; ++mf)
#pragma unroll
      for (int nf = 0; nf < 2; ++nf)
        acc[mf][nf] = __builtin_amdgcn_wmma_f32_16x16x32_bf16(
            false, a[mf].v, false, b[nf].v, (short)0, acc[mf][nf], false,
            false);

    __syncthreads();  // stage i reads done before buf is rewritten (stage i+2)
  }

  // Epilogue: C/D layout -> VGPR r is row r (lanes 0-15) / r+8 (lanes 16-31)
#pragma unroll
  for (int mf = 0; mf < 4; ++mf) {
#pragma unroll
    for (int nf = 0; nf < 2; ++nf) {
      int col     = n0 + wn * 32 + nf * 16 + lr;
      float bv    = bias[col];
      int rowbase = m0 + wm * 64 + mf * 16 + hi * 8;
#pragma unroll
      for (int r = 0; r < 8; ++r) {
        float v = acc[mf][nf][r] + bv;
        if (RELU) v = v > 0.0f ? v : 0.0f;
        size_t idx = (size_t)(rowbase + r) * N + col;
        if (OUTBF16)
          outb[idx] = (bf16)v;
        else
          outf[idx] = v;
      }
    }
  }
}

extern "C" void kernel_launch(void* const* d_in, const int* in_sizes, int n_in,
                              void* d_out, int out_size, void* d_ws,
                              size_t ws_size, hipStream_t stream) {
  constexpr int BATCH = 4096, DIMF = 4096, HID = 4096, LAB = 1024;

  const float* x  = (const float*)d_in[0];
  const float* w1 = (const float*)d_in[1];
  const float* b1 = (const float*)d_in[2];
  const float* w2 = (const float*)d_in[3];
  const float* b2 = (const float*)d_in[4];
  float* out = (float*)d_out;

  // Workspace: xb 32MB @0, w1b 32MB @32MB, w2b 8MB @64MB, h 32MB @72MB
  char* ws = (char*)d_ws;
  bf16* xb  = (bf16*)(ws);
  bf16* w1b = (bf16*)(ws + (size_t)33554432);
  bf16* w2b = (bf16*)(ws + (size_t)67108864);
  bf16* h   = (bf16*)(ws + (size_t)75497472);
  if (ws_size < (size_t)109051904) return;  // insufficient scratch

  const int nx4  = BATCH * DIMF / 4;
  const int nw14 = HID * DIMF / 4;
  const int nw24 = LAB * HID / 4;

  cvt_bf16_kernel<<<nx4 / 256, 256, 0, stream>>>(x, xb, nx4);
  binarize_bf16_kernel<<<nw14 / 256, 256, 0, stream>>>(w1, w1b, nw14);
  binarize_bf16_kernel<<<nw24 / 256, 256, 0, stream>>>(w2, w2b, nw24);

  dim3 g1(HID / BN, BATCH / BM);
  gemm_wmma_bf16<true, true><<<g1, 256, 0, stream>>>(
      xb, w1b, b1, h, nullptr, BATCH, HID, DIMF);

  dim3 g2(LAB / BN, BATCH / BM);
  gemm_wmma_bf16<false, false><<<g2, 256, 0, stream>>>(
      h, w2b, b2, nullptr, out, BATCH, LAB, HID);
}
